// GraphSAGE_PI_72181220377205
// MI455X (gfx1250) — compile-verified, matrix operand order checked by reading, not run on hardware
//
#include <hip/hip_runtime.h>
#include <hip/hip_bf16.h>

typedef __attribute__((ext_vector_type(16))) _Float16 v16h;
typedef __attribute__((ext_vector_type(8)))  float    v8f;

#define HID    128
#define LN_EPS 1e-5f

// ---------------------------------------------------------------------------
// zero fill
// ---------------------------------------------------------------------------
__global__ void zero_f32_kernel(float* __restrict__ p, long long count) {
    long long i = (long long)blockIdx.x * blockDim.x + threadIdx.x;
    long long stride = (long long)gridDim.x * blockDim.x;
    for (; i < count; i += stride) p[i] = 0.0f;
}

// ---------------------------------------------------------------------------
// degree count + invert
// ---------------------------------------------------------------------------
__global__ void degree_kernel(const int* __restrict__ dst, float* __restrict__ deg, int E) {
    int e = blockIdx.x * blockDim.x + threadIdx.x;
    if (e < E) atomicAdd(&deg[dst[e]], 1.0f);
}

__global__ void invert_deg_kernel(float* __restrict__ deg, int n) {
    int i = blockIdx.x * blockDim.x + threadIdx.x;
    if (i < n) deg[i] = 1.0f / fmaxf(deg[i], 1.0f);
}

// ---------------------------------------------------------------------------
// Pack a 128x128 row-major (K x N) f32 weight matrix into WMMA-B fragment
// order (f16).  For column-tile nt (8), k-step kt (4), lane (32), the 16
// f16 values a lane feeds to v_wmma_f32_16x16x32_f16 are stored contiguously:
//   ncol = nt*16 + lane%16,  kb = (lane/16)*8
//   j<8 : K = kt*32 + kb + j        j>=8 : K = kt*32 + kb + 16 + (j-8)
// ---------------------------------------------------------------------------
__global__ void pack_weight_kernel(const float* __restrict__ W, _Float16* __restrict__ Wp) {
    int gid = blockIdx.x * blockDim.x + threadIdx.x;   // 16384 = 128*128
    if (gid >= HID * HID) return;
    int j    = gid & 15;
    int lane = (gid >> 4) & 31;
    int kt   = (gid >> 9) & 3;
    int nt   = gid >> 11;
    int ncol = nt * 16 + (lane & 15);
    int kb   = (lane >> 4) * 8;
    int K    = kt * 32 + kb + ((j < 8) ? j : (8 + j)); // 16 + (j-8)
    Wp[gid] = (_Float16)W[K * HID + ncol];
}

// ---------------------------------------------------------------------------
// scatter-add: one wave per edge; each lane moves 4 contiguous floats.
// Coalesced 512B row read of x[src], coalesced f32 atomics into agg[dst].
// ---------------------------------------------------------------------------
__global__ void scatter_kernel(const float* __restrict__ x,
                               const int* __restrict__ src,
                               const int* __restrict__ dst,
                               float* __restrict__ agg, int E) {
    long long gid = (long long)blockIdx.x * blockDim.x + threadIdx.x;
    int e = (int)(gid >> 5);
    if (e >= E) return;
    int chunk = ((int)gid & 31) * 4;
    const float4 v = *(const float4*)(x + (long long)src[e] * HID + chunk);
    float* a = agg + (long long)dst[e] * HID + chunk;
    atomicAdd(a + 0, v.x);
    atomicAdd(a + 1, v.y);
    atomicAdd(a + 2, v.z);
    atomicAdd(a + 3, v.w);
}

// ---------------------------------------------------------------------------
// Load one 16x32 f16 A-fragment slice for this lane (no scaling):
// two 32B f32 chunks at K = base..base+7 and base+16..base+23, converted.
// ---------------------------------------------------------------------------
__device__ __forceinline__ v16h load_a_frag(const float* __restrict__ ap) {
    const float4 a0 = *(const float4*)(ap);
    const float4 a1 = *(const float4*)(ap + 4);
    const float4 a2 = *(const float4*)(ap + 16);
    const float4 a3 = *(const float4*)(ap + 20);
    v16h a;
    a[0]  = (_Float16)a0.x; a[1]  = (_Float16)a0.y;
    a[2]  = (_Float16)a0.z; a[3]  = (_Float16)a0.w;
    a[4]  = (_Float16)a1.x; a[5]  = (_Float16)a1.y;
    a[6]  = (_Float16)a1.z; a[7]  = (_Float16)a1.w;
    a[8]  = (_Float16)a2.x; a[9]  = (_Float16)a2.y;
    a[10] = (_Float16)a2.z; a[11] = (_Float16)a2.w;
    a[12] = (_Float16)a3.x; a[13] = (_Float16)a3.y;
    a[14] = (_Float16)a3.z; a[15] = (_Float16)a3.w;
    return a;
}

// ---------------------------------------------------------------------------
// Fused SAGE layer:  out = relu(LN( (agg@W_l)*deg_inv_row + b_l + xin@W_r ))
// (row-scaling by deg_inv commutes through the GEMM -> applied in epilogue)
// Block = 256 threads = 8 waves, owns 16 rows.  Wave w -> columns [16w,16w+16).
// Two independent WMMA accumulator chains hide the WMMA RAW-hazard slots.
// ---------------------------------------------------------------------------
__global__ void
__launch_bounds__(256)
fused_sage_kernel(const float* __restrict__ agg,
                  const float* __restrict__ deg_inv,
                  const float* __restrict__ xin,
                  const _Float16* __restrict__ Wp_l,
                  const float* __restrict__ b_l,
                  const _Float16* __restrict__ Wp_r,
                  const float* __restrict__ gamma,
                  const float* __restrict__ beta,
                  float* __restrict__ out) {
    __shared__ float lds[16 * HID];   // 8 KB f32 tile

    const int tid      = threadIdx.x;
    const int wave     = tid >> 5;            // 0..7
    const int lane     = tid & 31;            // wave32
    const int laneMod  = lane & 15;
    const int laneHalf = lane >> 4;
    const int rowBlock = blockIdx.x * 16;     // N = 3125*16 exactly, no tail
    const int n0       = wave * 16;

    const int   arow  = rowBlock + laneMod;
    const float scale = deg_inv[arow];        // lanes 0..15 hold rows 0..15
    const int   ncol  = n0 + laneMod;
    const int   kb    = laneHalf * 8;

    // per-wave packed-weight base: fragment index = (wave*4 + kt)*32 + lane
    const v16h* bl = (const v16h*)(Wp_l) + (wave * 4) * 32 + lane;
    const v16h* br = (const v16h*)(Wp_r) + (wave * 4) * 32 + lane;

    const float* arowp = agg + (long long)arow * HID;
    const float* xrowp = xin + (long long)arow * HID;

    v8f accL = {};   // agg @ W_l (unscaled)
    v8f accR = {};   // xin @ W_r

#pragma unroll
    for (int kt = 0; kt < 4; ++kt) {
        const v16h aL = load_a_frag(arowp + kt * 32 + kb);
        const v16h aR = load_a_frag(xrowp + kt * 32 + kb);
        const v16h bL = bl[kt * 32];      // one coalesced 32B load per lane
        const v16h bR = br[kt * 32];
        accL = __builtin_amdgcn_wmma_f32_16x16x32_f16(
            false, aL, false, bL, (short)0, accL, false, false);
        accR = __builtin_amdgcn_wmma_f32_16x16x32_f16(
            false, aR, false, bR, (short)0, accR, false, false);
    }

    // ---- epilogue: per-row deg_inv via shuffle, +bias, spill to LDS ----
    // C layout: vgpr i, laneHalf h -> row M = i + 8h; that row's deg_inv
    // lives in lane (i + 8h) as `scale`.
    const float bias = b_l[ncol];
#pragma unroll
    for (int i = 0; i < 8; ++i) {
        const float d = __shfl(scale, i + 8 * laneHalf, 32);
        lds[(i + 8 * laneHalf) * HID + ncol] = fmaf(accL[i], d, accR[i]) + bias;
    }
    __syncthreads();

    // ---- LayerNorm + ReLU: each wave handles 2 of the 16 rows ----
#pragma unroll
    for (int rr = 0; rr < 2; ++rr) {
        const int r   = wave * 2 + rr;
        const int col = lane * 4;
        const float4 v = *(const float4*)(&lds[r * HID + col]);
        float s  = v.x + v.y + v.z + v.w;
        float sq = v.x * v.x + v.y * v.y + v.z * v.z + v.w * v.w;
#pragma unroll
        for (int off = 16; off > 0; off >>= 1) {
            s  += __shfl_xor(s,  off, 32);
            sq += __shfl_xor(sq, off, 32);
        }
        const float mean = s * (1.0f / HID);
        const float var  = sq * (1.0f / HID) - mean * mean;
        const float inv  = rsqrtf(var + LN_EPS);
        const float4 g  = *(const float4*)(gamma + col);
        const float4 bb = *(const float4*)(beta + col);
        float4 o;
        o.x = fmaxf((v.x - mean) * inv * g.x + bb.x, 0.0f);
        o.y = fmaxf((v.y - mean) * inv * g.y + bb.y, 0.0f);
        o.z = fmaxf((v.z - mean) * inv * g.z + bb.z, 0.0f);
        o.w = fmaxf((v.w - mean) * inv * g.w + bb.w, 0.0f);
        *(float4*)(out + (long long)(rowBlock + r) * HID + col) = o;
    }
}

// ---------------------------------------------------------------------------
// head: out[n,o] = sum_k h[n,k] * W_h[k,o] + b_h[o]   (OUT = 4)
// ---------------------------------------------------------------------------
__global__ void head_kernel(const float* __restrict__ h,
                            const float* __restrict__ W_h,
                            const float* __restrict__ b_h,
                            float* __restrict__ out, int n_nodes) {
    int gid = blockIdx.x * blockDim.x + threadIdx.x;
    if (gid >= n_nodes * 4) return;
    int n = gid >> 2;
    int o = gid & 3;
    const float* hr = h + (long long)n * HID;
    float s = b_h[o];
#pragma unroll 8
    for (int k = 0; k < HID; ++k) s += hr[k] * W_h[k * 4 + o];
    out[gid] = s;
}

// ---------------------------------------------------------------------------
extern "C" void kernel_launch(void* const* d_in, const int* in_sizes, int n_in,
                              void* d_out, int out_size, void* d_ws, size_t ws_size,
                              hipStream_t stream) {
    const float* x     = (const float*)d_in[0];
    const int*   ei    = (const int*)d_in[1];     // [2, E] : src then dst
    const float* W_l0  = (const float*)d_in[2];
    const float* b_l0  = (const float*)d_in[3];
    const float* W_r0  = (const float*)d_in[4];
    const float* W_l1  = (const float*)d_in[5];
    const float* b_l1  = (const float*)d_in[6];
    const float* W_r1  = (const float*)d_in[7];
    const float* g0    = (const float*)d_in[8];
    const float* beta0 = (const float*)d_in[9];
    const float* g1    = (const float*)d_in[10];
    const float* beta1 = (const float*)d_in[11];
    const float* W_h   = (const float*)d_in[12];
    const float* b_h   = (const float*)d_in[13];

    const int N_ = in_sizes[0] / HID;     // 50000
    const int E_ = in_sizes[1] / 2;       // 800000
    const int* src = ei;
    const int* dst = ei + E_;

    // workspace: agg [N*128 f32] | h [N*128 f32] | deg_inv [N f32] |
    //            4 packed f16 weights [128*128 each]          (~51.6 MB)
    float*    agg    = (float*)d_ws;
    float*    hbuf   = agg + (long long)N_ * HID;
    float*    deginv = hbuf + (long long)N_ * HID;
    _Float16* wp     = (_Float16*)(deginv + N_);
    _Float16* Wp_l0  = wp;
    _Float16* Wp_r0  = wp + 1 * HID * HID;
    _Float16* Wp_l1  = wp + 2 * HID * HID;
    _Float16* Wp_r1  = wp + 3 * HID * HID;

    const long long NH = (long long)N_ * HID;
    const int ZB = 1024;

    // ---- prep: degree->inv, pack weights into WMMA-B fragment order ----
    zero_f32_kernel<<<ZB, 256, 0, stream>>>(deginv, N_);
    degree_kernel<<<(E_ + 255) / 256, 256, 0, stream>>>(dst, deginv, E_);
    invert_deg_kernel<<<(N_ + 255) / 256, 256, 0, stream>>>(deginv, N_);
    pack_weight_kernel<<<64, 256, 0, stream>>>(W_l0, Wp_l0);
    pack_weight_kernel<<<64, 256, 0, stream>>>(W_r0, Wp_r0);
    pack_weight_kernel<<<64, 256, 0, stream>>>(W_l1, Wp_l1);
    pack_weight_kernel<<<64, 256, 0, stream>>>(W_r1, Wp_r1);

    const long long scatThreads = (long long)E_ * 32;
    const int scatBlocks = (int)((scatThreads + 255) / 256);
    const int rowBlocks  = N_ / 16;       // 3125, exact

    // ---- layer 0 ----
    zero_f32_kernel<<<ZB, 256, 0, stream>>>(agg, NH);
    scatter_kernel<<<scatBlocks, 256, 0, stream>>>(x, src, dst, agg, E_);
    fused_sage_kernel<<<rowBlocks, 256, 0, stream>>>(
        agg, deginv, x, Wp_l0, b_l0, Wp_r0, g0, beta0, hbuf);

    // ---- layer 1 ----
    zero_f32_kernel<<<ZB, 256, 0, stream>>>(agg, NH);
    scatter_kernel<<<scatBlocks, 256, 0, stream>>>(hbuf, src, dst, agg, E_);
    // output aliases agg: each block reads only its own 16 rows (pre-barrier)
    // and writes the same 16 rows (post-barrier) -> safe.
    fused_sage_kernel<<<rowBlocks, 256, 0, stream>>>(
        agg, deginv, hbuf, Wp_l1, b_l1, Wp_r1, g1, beta1, agg);

    // ---- head ----
    head_kernel<<<(N_ * 4 + 255) / 256, 256, 0, stream>>>(
        agg, W_h, b_h, (float*)d_out, N_);
}
